// ThreeLayerGraphTransformer_29094108463691
// MI455X (gfx1250) — compile-verified, matrix operand order checked by reading, not run on hardware
//
#include <hip/hip_runtime.h>
#include <hip/hip_bf16.h>
#include <math.h>

#define NNODES 10000
#define NEDGES 160000
#define DINDIM 128
#define HDIM   256
#define DEDIM  32

typedef __attribute__((ext_vector_type(16))) __bf16 v16bf;
typedef __attribute__((ext_vector_type(8)))  float  v8f;
typedef __attribute__((ext_vector_type(4)))  unsigned int u32x4;
typedef unsigned short u16;

// ---------- helpers ----------
__device__ __forceinline__ u16 f2bf(float f) {
  unsigned u = __float_as_uint(f);
  u += 0x7FFFu + ((u >> 16) & 1u);   // round-to-nearest-even
  return (u16)(u >> 16);
}
__device__ __forceinline__ float bf2f(u16 h) {
  return __uint_as_float(((unsigned)h) << 16);
}
__device__ __forceinline__ void atomicMaxF(float* addr, float v) {
  if (v >= 0.0f) atomicMax((int*)addr, __float_as_int(v));
  else           atomicMin((unsigned int*)addr, __float_as_uint(v));
}

// ---------- conversions ----------
__global__ void cvt_f32_bf16(const float* __restrict__ in, u16* __restrict__ out, int n) {
  int i = blockIdx.x * blockDim.x + threadIdx.x;
  if (i < n) out[i] = f2bf(in[i]);
}

// Convert + swizzle a weight matrix [K,NOUT] f32 into WMMA-fragment order:
// flat index = ((((nb*(K/32) + kt)*2 + hi)*128 + c)*16 + i
// so a GEMM block's B tile is one contiguous region, and each lane's
// 16-element fragment is 32 contiguous bytes in LDS.
template <int K, int NOUT>
__global__ void swizzle_w(const float* __restrict__ W, u16* __restrict__ out)
{
  int t = blockIdx.x * blockDim.x + threadIdx.x;
  if (t >= K * NOUT) return;
  int i  = t & 15;
  int c  = (t >> 4) & 127;
  int hi = (t >> 11) & 1;
  int r  = t >> 12;
  int kt = r % (K / 32);
  int nb = r / (K / 32);
  int k  = kt * 32 + hi * 8 + ((i < 8) ? i : i + 8);   // 16-bit A/B 16x32 K map
  out[t] = f2bf(W[(size_t)k * NOUT + nb * 128 + c]);
}

// ---------- bf16 WMMA GEMM ----------
// out[M,NOUT] = A[M,K](bf16) * W(swizzled bf16) (+bias)(relu)
// block = 256 threads = 8 waves; tile = 16(M) x 128(N); wave owns 16 cols.
// Requires M % 16 == 0 (true for all launches: 10000, 160000).
template <int K, int NOUT, bool BIAS, bool RELU, bool OUTF>
__global__ __launch_bounds__(256)
void gemm_tile(const u16* __restrict__ A, const u16* __restrict__ Wsw,
               const float* __restrict__ bias,
               float* __restrict__ outF, u16* __restrict__ outB)
{
  constexpr int K8 = K / 8;                  // uint4 per A row
  __shared__ u32x4 lsA4[16 * K8];            // 16 x K bf16 (row-major)
  __shared__ u32x4 lsB4[K * 16];             // K x 128 bf16 (fragment-major)

  const int tid  = threadIdx.x;
  const int wave = tid >> 5;
  const int lane = tid & 31;
  const int hi   = lane >> 4;
  const int l16  = lane & 15;
  const int m0   = blockIdx.x * 16;
  const int nb   = blockIdx.y;
  const int colg = nb * 128 + wave * 16 + l16;

  // stage A tile: coalesced b128
  #pragma unroll
  for (int idx = tid; idx < 16 * K8; idx += 256) {
    int r = idx / K8, c = idx - r * K8;
    lsA4[idx] = ((const u32x4*)A)[(size_t)(m0 + r) * K8 + c];
  }
  // stage B tile: contiguous pre-swizzled region, coalesced b128
  const u32x4* Wblk = (const u32x4*)Wsw + (size_t)nb * (K * 16);
  #pragma unroll
  for (int idx = tid; idx < K * 16; idx += 256) {
    lsB4[idx] = Wblk[idx];
  }
  __syncthreads();

  v8f acc = {};
  #pragma unroll
  for (int kt = 0; kt < K / 32; ++kt) {
    union { v16bf v; u32x4 q[2]; } a, b;
    // A fragment: two contiguous 16B chunks (k = kt*32+hi*8+{0..7}, +16)
    int abase = (l16 * K + kt * 32 + hi * 8) >> 3;
    a.q[0] = lsA4[abase];
    a.q[1] = lsA4[abase + 2];
    // B fragment: 32 contiguous bytes in fragment-major layout
    int bbase = ((kt * 2 + hi) * 128 + wave * 16 + l16) * 2;
    b.q[0] = lsB4[bbase];
    b.q[1] = lsB4[bbase + 1];
    acc = __builtin_amdgcn_wmma_f32_16x16x32_bf16(false, a.v, false, b.v,
                                                  (short)0, acc, false, false);
  }

  const float bv = BIAS ? bias[colg] : 0.0f;
  const size_t base = (size_t)(m0 + hi * 8) * NOUT + colg;
  #pragma unroll
  for (int j = 0; j < 8; ++j) {
    float r = acc[j] + bv;
    if (RELU) r = fmaxf(r, 0.0f);
    if (OUTF) outF[base + (size_t)j * NOUT] = r;
    else      outB[base + (size_t)j * NOUT] = f2bf(r);
  }
}

// ---------- attention phase ----------
__global__ void init_node(float* m, float* denom, int n) {
  int i = blockIdx.x * blockDim.x + threadIdx.x;
  if (i < n) { m[i] = -__builtin_inff(); denom[i] = 0.0f; }
}
__global__ void zero_f32(float* p, int n) {
  int i = blockIdx.x * blockDim.x + threadIdx.x;
  if (i < n) p[i] = 0.0f;
}

// one wave per edge: alpha = dot(q[dst], k[src]+e) * scale; atomic segment-max
__global__ __launch_bounds__(256)
void edge_alpha(const float* __restrict__ q, const float* __restrict__ k,
                const u16* __restrict__ ebf, const int* __restrict__ src,
                const int* __restrict__ dst, float* __restrict__ alpha,
                float* __restrict__ m, int E, float scale)
{
  int wid  = (blockIdx.x * blockDim.x + threadIdx.x) >> 5;
  int lane = threadIdx.x & 31;
  if (wid >= E) return;
  int s = src[wid], d = dst[wid];
  const float* qr = q + (size_t)d * HDIM;
  const float* kr = k + (size_t)s * HDIM;
  const u16*   er = ebf + (size_t)wid * HDIM;
  float sum = 0.0f;
#pragma unroll
  for (int j = 0; j < HDIM / 32; ++j) {
    int c = j * 32 + lane;
    sum += qr[c] * (kr[c] + bf2f(er[c]));
  }
#pragma unroll
  for (int off = 16; off >= 1; off >>= 1) sum += __shfl_xor(sum, off, 32);
  if (lane == 0) {
    float a = sum * scale;
    alpha[wid] = a;
    atomicMaxF(&m[d], a);
  }
}

__global__ void edge_exp(const float* __restrict__ alpha, const int* __restrict__ dst,
                         const float* __restrict__ m, float* __restrict__ aexp,
                         float* __restrict__ denom, int E)
{
  int i = blockIdx.x * blockDim.x + threadIdx.x;
  if (i < E) {
    int d = dst[i];
    float a = __expf(alpha[i] - m[d]);
    aexp[i] = a;
    atomicAdd(&denom[d], a);
  }
}

// one wave per edge: acc[dst] += (v[src] + e) * a
__global__ __launch_bounds__(256)
void edge_message(const float* __restrict__ v, const u16* __restrict__ ebf,
                  const int* __restrict__ src, const int* __restrict__ dst,
                  const float* __restrict__ aexp, const float* __restrict__ denom,
                  float* __restrict__ acc, int E)
{
  int wid  = (blockIdx.x * blockDim.x + threadIdx.x) >> 5;
  int lane = threadIdx.x & 31;
  if (wid >= E) return;
  int s = src[wid], d = dst[wid];
  float coeff = aexp[wid] / fmaxf(denom[d], 1e-16f);
  const float* vr = v + (size_t)s * HDIM;
  const u16*   er = ebf + (size_t)wid * HDIM;
  float*       ar = acc + (size_t)d * HDIM;
#pragma unroll
  for (int j = 0; j < HDIM / 32; ++j) {
    int c = j * 32 + lane;
    atomicAdd(&ar[c], (vr[c] + bf2f(er[c])) * coeff);
  }
}

template <bool RELU, bool OUTF>
__global__ void finalize(const float* __restrict__ acc, const float* __restrict__ s,
                         float* __restrict__ outF, u16* __restrict__ outB, int n)
{
  int i = blockIdx.x * blockDim.x + threadIdx.x;
  if (i < n) {
    float r = acc[i] + s[i];
    if (RELU) r = fmaxf(r, 0.0f);
    if (OUTF) outF[i] = r;
    else      outB[i] = f2bf(r);
  }
}

// ---------- host orchestration ----------
extern "C" void kernel_launch(void* const* d_in, const int* in_sizes, int n_in,
                              void* d_out, int out_size, void* d_ws, size_t ws_size,
                              hipStream_t stream)
{
  const float* x         = (const float*)d_in[0];
  const float* edge_attr = (const float*)d_in[1];
  const int*   src       = (const int*)d_in[2];
  const int*   dst       = (const int*)d_in[3];
  const float* W_edge    = (const float*)d_in[4];
  const float* b_edge    = (const float*)d_in[5];
  const float* W_node    = (const float*)d_in[6];
  const float* b_node    = (const float*)d_in[7];
  const float *Wq[3], *bq[3], *Wk[3], *bk[3], *Wv[3], *bv[3], *We[3], *Wss[3], *bs[3];
  for (int l = 0; l < 3; ++l) {
    int base = 8 + l * 9;
    Wq[l]  = (const float*)d_in[base + 0]; bq[l] = (const float*)d_in[base + 1];
    Wk[l]  = (const float*)d_in[base + 2]; bk[l] = (const float*)d_in[base + 3];
    Wv[l]  = (const float*)d_in[base + 4]; bv[l] = (const float*)d_in[base + 5];
    We[l]  = (const float*)d_in[base + 6];
    Wss[l] = (const float*)d_in[base + 7]; bs[l] = (const float*)d_in[base + 8];
  }

  // ---- workspace carve (256B aligned) ----
  char* wp = (char*)d_ws;
  auto carve = [&](size_t bytes) -> void* {
    void* p = (void*)wp;
    wp += (bytes + 255) & ~(size_t)255;
    return p;
  };
  u16*   xbf   = (u16*)carve((size_t)NNODES * DINDIM * 2);
  u16*   eabf  = (u16*)carve((size_t)NEDGES * DEDIM * 2);
  u16*   efbf  = (u16*)carve((size_t)NEDGES * HDIM * 2);   // edge_feat (bf16)
  u16*   hbf   = (u16*)carve((size_t)NNODES * HDIM * 2);   // h (bf16)
  u16*   ebf   = (u16*)carve((size_t)NEDGES * HDIM * 2);   // per-layer e (bf16)
  float* q     = (float*)carve((size_t)NNODES * HDIM * 4);
  float* k     = (float*)carve((size_t)NNODES * HDIM * 4);
  float* v     = (float*)carve((size_t)NNODES * HDIM * 4);
  float* sbuf  = (float*)carve((size_t)NNODES * HDIM * 4); // skip h@Ws+bs
  float* acc   = (float*)carve((size_t)NNODES * HDIM * 4);
  float* alpha = (float*)carve((size_t)NEDGES * 4);
  float* aexp  = (float*)carve((size_t)NEDGES * 4);
  float* mmax  = (float*)carve((size_t)NNODES * 4);
  float* denom = (float*)carve((size_t)NNODES * 4);
  // swizzled bf16 weights (L2-resident, ~2MB total)
  u16* WedgeS = (u16*)carve((size_t)DEDIM * HDIM * 2);
  u16* WnodeS = (u16*)carve((size_t)DINDIM * HDIM * 2);
  u16 *WqS[3], *WkS[3], *WvS[3], *WeS[3], *WsS[3];
  for (int l = 0; l < 3; ++l) {
    WqS[l] = (u16*)carve((size_t)HDIM * HDIM * 2);
    WkS[l] = (u16*)carve((size_t)HDIM * HDIM * 2);
    WvS[l] = (u16*)carve((size_t)HDIM * HDIM * 2);
    WeS[l] = (u16*)carve((size_t)HDIM * HDIM * 2);
    WsS[l] = (u16*)carve((size_t)HDIM * HDIM * 2);
  }

  // activations -> bf16
  cvt_f32_bf16<<<(NNODES * DINDIM + 255) / 256, 256, 0, stream>>>(x, xbf, NNODES * DINDIM);
  cvt_f32_bf16<<<(NEDGES * DEDIM + 255) / 256, 256, 0, stream>>>(edge_attr, eabf, NEDGES * DEDIM);

  // weights -> bf16 + WMMA-fragment swizzle
  swizzle_w<DEDIM,  HDIM><<<(DEDIM  * HDIM + 255) / 256, 256, 0, stream>>>(W_edge, WedgeS);
  swizzle_w<DINDIM, HDIM><<<(DINDIM * HDIM + 255) / 256, 256, 0, stream>>>(W_node, WnodeS);
  for (int l = 0; l < 3; ++l) {
    swizzle_w<HDIM, HDIM><<<(HDIM * HDIM + 255) / 256, 256, 0, stream>>>(Wq[l],  WqS[l]);
    swizzle_w<HDIM, HDIM><<<(HDIM * HDIM + 255) / 256, 256, 0, stream>>>(Wk[l],  WkS[l]);
    swizzle_w<HDIM, HDIM><<<(HDIM * HDIM + 255) / 256, 256, 0, stream>>>(Wv[l],  WvS[l]);
    swizzle_w<HDIM, HDIM><<<(HDIM * HDIM + 255) / 256, 256, 0, stream>>>(We[l],  WeS[l]);
    swizzle_w<HDIM, HDIM><<<(HDIM * HDIM + 255) / 256, 256, 0, stream>>>(Wss[l], WsS[l]);
  }

  const dim3 blk(256);
  const dim3 gN(NNODES / 16, HDIM / 128);   // 625 x 2
  const dim3 gE(NEDGES / 16, HDIM / 128);   // 10000 x 2

  // encoders: relu, bf16 out
  gemm_tile<DEDIM,  HDIM, true, true, false><<<gE, blk, 0, stream>>>(
      eabf, WedgeS, b_edge, nullptr, efbf);
  gemm_tile<DINDIM, HDIM, true, true, false><<<gN, blk, 0, stream>>>(
      xbf, WnodeS, b_node, nullptr, hbf);

  const float scale = 0.0625f;  // 1/sqrt(256)
  for (int l = 0; l < 3; ++l) {
    // q,k,v,skip: bias, no relu, f32 out
    gemm_tile<HDIM, HDIM, true, false, true><<<gN, blk, 0, stream>>>(
        hbf, WqS[l], bq[l], q, nullptr);
    gemm_tile<HDIM, HDIM, true, false, true><<<gN, blk, 0, stream>>>(
        hbf, WkS[l], bk[l], k, nullptr);
    gemm_tile<HDIM, HDIM, true, false, true><<<gN, blk, 0, stream>>>(
        hbf, WvS[l], bv[l], v, nullptr);
    gemm_tile<HDIM, HDIM, true, false, true><<<gN, blk, 0, stream>>>(
        hbf, WsS[l], bs[l], sbuf, nullptr);
    // e: no bias, no relu, bf16 out
    gemm_tile<HDIM, HDIM, false, false, false><<<gE, blk, 0, stream>>>(
        efbf, WeS[l], nullptr, nullptr, ebf);

    init_node<<<(NNODES + 255) / 256, 256, 0, stream>>>(mmax, denom, NNODES);
    zero_f32<<<(NNODES * HDIM + 255) / 256, 256, 0, stream>>>(acc, NNODES * HDIM);

    edge_alpha<<<(NEDGES * 32 + 255) / 256, 256, 0, stream>>>(
        q, k, ebf, src, dst, alpha, mmax, NEDGES, scale);
    edge_exp<<<(NEDGES + 255) / 256, 256, 0, stream>>>(
        alpha, dst, mmax, aexp, denom, NEDGES);
    edge_message<<<(NEDGES * 32 + 255) / 256, 256, 0, stream>>>(
        v, ebf, src, dst, aexp, denom, acc, NEDGES);

    if (l == 2) {
      finalize<false, true><<<(NNODES * HDIM + 255) / 256, 256, 0, stream>>>(
          acc, sbuf, (float*)d_out, nullptr, NNODES * HDIM);
    } else {
      finalize<true, false><<<(NNODES * HDIM + 255) / 256, 256, 0, stream>>>(
          acc, sbuf, nullptr, hbf, NNODES * HDIM);
    }
  }
}
